// TransformerLayers_58162447123013
// MI455X (gfx1250) — matrix operand order repaired and checked by reading, not compiled, not run on hardware
//
#include <hip/hip_runtime.h>
#include <math.h>

#define Bb   2
#define Nn   8192
#define Cc   32
#define Kk   16
#define Rr   8
#define Nss  (Nn / Rr)           // 1024
#define Hh   260
#define Ww   346
#define FSs  5
#define CIN2 (Cc + 2)            // 34
#define KCONV (FSs * FSs * CIN2) // 850
#define KCONVP 864               // padded to multiple of 32
#define EPSf 1e-5f
#define INV_SCALE 0.17677669529663687f  // 1/sqrt(32)
#define CHUNK 256                // points staged to LDS per KNN chunk

typedef __attribute__((ext_vector_type(16))) _Float16 v16h;
typedef __attribute__((ext_vector_type(8)))  float    v8f;
typedef __attribute__((ext_vector_type(4)))  float    f32x4;

// ---------------- helpers ----------------
__device__ __forceinline__ float wave_sum(float v) {
  #pragma unroll
  for (int o = 16; o; o >>= 1) v += __shfl_xor(v, o, 32);
  return v;
}

__global__ void fill_kernel(float* p, float v, size_t n) {
  size_t i = (size_t)blockIdx.x * blockDim.x + threadIdx.x;
  if (i < n) p[i] = v;
}

// ---------------- weight pre-swizzle into f16 B-fragment layout ----------------
// Wswz[ks][nt][lane][e] -> element (K = ks*32 + 8*(lane>=16) + e + 8*(e>=8),
//                                   N = nt*16 + (lane&15))
__global__ void wswz_kernel(const float* __restrict__ W, _Float16* __restrict__ Wswz,
                            int Kd, int Nd) {
  const int ksteps = (Kd + 31) >> 5;
  const int ntiles = Nd >> 4;
  const int t = blockIdx.x * blockDim.x + threadIdx.x;
  if (t >= ksteps * ntiles * 32) return;
  const int lane = t & 31;
  const int nt = (t >> 5) % ntiles;
  const int ks = t / (32 * ntiles);
  const int g = lane >> 4;
  const int col = nt * 16 + (lane & 15);
  _Float16* dst = Wswz + ((size_t)(ks * ntiles + nt) * 32 + lane) * 16;
  #pragma unroll
  for (int e = 0; e < 16; ++e) {
    const int kk = ks * 32 + 8 * g + e + (e >= 8 ? 8 : 0);
    dst[e] = (_Float16)((kk < Kd) ? W[(size_t)kk * Nd + col] : 0.f);
  }
}

// ---------------- WMMA GEMM: Out[M, NT*16] = A[M,Kd] * W + bias ----------------
// One wave computes a 16-row strip across all NT column tiles. Kd % 32 == 0,
// M % 16 == 0: no bounds checks anywhere in the hot loop.
template <int NT>
__global__ void gemm_wmma_kernel(const float* __restrict__ A,
                                 const _Float16* __restrict__ Wswz,
                                 const float* __restrict__ bias,
                                 float* __restrict__ Out, int M, int Kd) {
  const int wave = threadIdx.x >> 5;
  const int lane = threadIdx.x & 31;
  const int strip = blockIdx.x * (blockDim.x >> 5) + wave;
  if (strip * 16 >= M) return;
  const int g  = lane >> 4;
  const int mr = lane & 15;
  const int row = strip * 16 + mr;

  v8f acc[NT];
  #pragma unroll
  for (int nt = 0; nt < NT; ++nt) {
    const float bv = bias ? bias[nt * 16 + mr] : 0.f;
    #pragma unroll
    for (int v = 0; v < 8; ++v) acc[nt][v] = bv;
  }

  const f32x4* Arow = (const f32x4*)(A + (size_t)row * Kd);
  const int ksteps = Kd >> 5;
  for (int ks = 0; ks < ksteps; ++ks) {
    const int base = ks * 8 + g * 2;        // float4 units
    const f32x4 c0 = Arow[base + 0];        // K = kb+8g   .. +3
    const f32x4 c1 = Arow[base + 1];        // K = kb+8g+4 .. +7
    const f32x4 c2 = Arow[base + 4];        // K = kb+16+8g   .. +3
    const f32x4 c3 = Arow[base + 5];        // K = kb+16+8g+4 .. +7
    v16h a;
    #pragma unroll
    for (int i = 0; i < 4; ++i) {
      a[i]      = (_Float16)c0[i];
      a[4 + i]  = (_Float16)c1[i];
      a[8 + i]  = (_Float16)c2[i];
      a[12 + i] = (_Float16)c3[i];
    }
    #pragma unroll
    for (int nt = 0; nt < NT; ++nt) {
      const v16h bf = *(const v16h*)(Wswz + ((size_t)(ks * NT + nt) * 32 + lane) * 16);
      acc[nt] = __builtin_amdgcn_wmma_f32_16x16x32_f16(false, a, false, bf,
                                                       (short)0, acc[nt], false, false);
    }
  }
  #pragma unroll
  for (int nt = 0; nt < NT; ++nt)
    #pragma unroll
    for (int v = 0; v < 8; ++v)
      Out[(size_t)(strip * 16 + v + 8 * g) * (NT * 16) + nt * 16 + mr] = acc[nt][v];
}

// ---------------- branchless register top-16 insert ----------------
__device__ __forceinline__ void top16_insert(float d, int m, float (&bd)[Kk], int (&bi)[Kk]) {
  if (d < bd[Kk - 1]) {
    float dd = d; int ii = m;
    #pragma unroll
    for (int j = 0; j < Kk; ++j) {
      if (dd < bd[j]) {
        float tf = bd[j]; bd[j] = dd; dd = tf;
        int   ti = bi[j]; bi[j] = ii; ii = ti;
      }
    }
  }
}

// ---------------- KNN: LDS-chunked scan with async global->LDS staging ----------------
// Requires: Nq % 256 == 0 (block uniform batch), Np % CHUNK == 0, grid = B*Nq/256.
__global__ void knn_kernel(const float* __restrict__ Q, int Nq,
                           const float* __restrict__ P, int Np,
                           int* __restrict__ outIdx) {
  __shared__ f32x4 sp[CHUNK];
  const int tid = threadIdx.x;
  const int t = blockIdx.x * blockDim.x + tid;
  const int b = t / Nq, n = t % Nq;
  const f32x4 q = *(const f32x4*)(Q + ((size_t)b * Nq + n) * 4);
  float bd[Kk]; int bi[Kk];
  #pragma unroll
  for (int k = 0; k < Kk; ++k) { bd[k] = 1e30f; bi[k] = 0; }
  const float* pb = P + (size_t)b * Np * 4;

  for (int base = 0; base < Np; base += CHUNK) {
    // async copy CHUNK points (one float4 per thread) into LDS
    {
      const unsigned lds = (unsigned)(uintptr_t)&sp[tid];
      const unsigned long long ga =
          (unsigned long long)(uintptr_t)(pb + (size_t)(base + tid) * 4);
      asm volatile("global_load_async_to_lds_b128 %0, %1, off"
                   :: "v"(lds), "v"(ga) : "memory");
      asm volatile("s_wait_asynccnt 0x0" ::: "memory");
    }
    __syncthreads();
    #pragma unroll 4
    for (int j = 0; j < CHUNK; ++j) {
      const f32x4 p = sp[j];
      const float dx = q[0] - p[0];
      const float dy = q[1] - p[1];
      const float dz = q[2] - p[2];
      top16_insert(dx * dx + dy * dy + dz * dz, base + j, bd, bi);
    }
    __syncthreads();
  }
  #pragma unroll
  for (int k = 0; k < Kk; ++k) outIdx[(size_t)t * Kk + k] = bi[k];
}

// ---------------- ball query on (comp1, comp2), first-K-by-index, pad with self ----------------
__global__ void ball_query_kernel(const float* __restrict__ xytp, float r2,
                                  int* __restrict__ outIdx) {
  const int t = blockIdx.x * blockDim.x + threadIdx.x;
  if (t >= Bb * Nn) return;
  const int b = t / Nn, n = t % Nn;
  const f32x4* pb = (const f32x4*)(xytp + (size_t)b * Nn * 4);
  const f32x4 q = pb[n];
  int cnt = 0;
  int idx[Kk];
  for (int m = 0; m < Nn && cnt < Kk; ++m) {
    const f32x4 p = pb[m];
    const float dy = q[1] - p[1];
    const float dz = q[2] - p[2];
    if (dy * dy + dz * dz <= r2) idx[cnt++] = m;
  }
  for (int k = cnt; k < Kk; ++k) idx[k] = n;
  #pragma unroll
  for (int k = 0; k < Kk; ++k) outIdx[(size_t)t * Kk + k] = idx[k];
}

// ---------------- farthest point sampling: one block per batch ----------------
__global__ void fps_kernel(const float* __restrict__ xytp, float* __restrict__ mind,
                           int* __restrict__ sidx) {
  __shared__ float smax[256];
  __shared__ int   simax[256];
  __shared__ int   s_last;
  const int b = blockIdx.x, tid = threadIdx.x;
  const f32x4* pts = (const f32x4*)(xytp + (size_t)b * Nn * 4);
  for (int n = tid; n < Nn; n += 256) mind[(size_t)b * Nn + n] = 1e30f;
  if (tid == 0) { sidx[b * Nss] = 0; s_last = 0; }
  __syncthreads();
  for (int it = 1; it < Nss; ++it) {
    const f32x4 lp = pts[s_last];
    float bm = -1.f; int bmi = 0;
    for (int n = tid; n < Nn; n += 256) {
      const f32x4 p = pts[n];
      const float dx = p[0] - lp[0], dy = p[1] - lp[1], dz = p[2] - lp[2];
      float md = mind[(size_t)b * Nn + n];
      md = fminf(md, dx * dx + dy * dy + dz * dz);
      mind[(size_t)b * Nn + n] = md;
      if (md > bm) { bm = md; bmi = n; }
    }
    smax[tid] = bm; simax[tid] = bmi;
    __syncthreads();
    for (int s = 128; s; s >>= 1) {
      if (tid < s) {
        if (smax[tid + s] > smax[tid] ||
            (smax[tid + s] == smax[tid] && simax[tid + s] < simax[tid])) {
          smax[tid] = smax[tid + s]; simax[tid] = simax[tid + s];
        }
      }
      __syncthreads();
    }
    if (tid == 0) { sidx[b * Nss + it] = simax[0]; s_last = simax[0]; }
    __syncthreads();
  }
}

__global__ void gather_sxytp_kernel(const float* __restrict__ xytp,
                                    const int* __restrict__ sidx,
                                    float* __restrict__ s_xytp) {
  const int t = blockIdx.x * blockDim.x + threadIdx.x;
  if (t >= Bb * Nss) return;
  const int b = t / Nss;
  const int m = sidx[t];
  *((f32x4*)s_xytp + t) = *((const f32x4*)xytp + (size_t)b * Nn + m);
}

// ---------------- pos-enc stage 1: x = rel @ w1 + b1, accumulate channel stats ----------------
__global__ void pe_stage1_kernel(const float* __restrict__ xytp,
                                 const float* __restrict__ src2, int srcN,
                                 const int* __restrict__ idx,
                                 const float* __restrict__ w1, const float* __restrict__ b1,
                                 float* __restrict__ xout, float* __restrict__ stats,
                                 int mode, int cin) {
  __shared__ float acc[8];
  const int t = blockIdx.x * blockDim.x + threadIdx.x;
  const bool active = t < Bb * Nn * Kk;
  if (threadIdx.x < 8) acc[threadIdx.x] = 0.f;
  __syncthreads();
  if (active) {
    const int b = t / (Nn * Kk);
    const int n = (t / Kk) % Nn;
    const int m = idx[t];
    float rel[4] = {0.f, 0.f, 0.f, 0.f};
    const f32x4 qp = *((const f32x4*)xytp + (size_t)b * Nn + n);
    if (mode == 0) {
      const f32x4 kp = *((const f32x4*)xytp + (size_t)b * Nn + m);
      #pragma unroll
      for (int i = 0; i < 4; ++i) rel[i] = qp[i] - kp[i];
    } else if (mode == 1) {
      const f32x4 kp = *((const f32x4*)xytp + (size_t)b * Nn + m);
      rel[0] = qp[1] - kp[1];
      rel[1] = qp[2] - kp[2];
    } else {
      const f32x4 kp = *((const f32x4*)src2 + (size_t)b * srcN + m);
      #pragma unroll
      for (int i = 0; i < 4; ++i) rel[i] = qp[i] - kp[i];
    }
    for (int j = 0; j < cin; ++j) {
      float x = b1[j];
      for (int i = 0; i < cin; ++i) x += rel[i] * w1[i * cin + j];
      xout[(size_t)t * cin + j] = x;
      atomicAdd(&acc[j], x);
      atomicAdd(&acc[4 + j], x * x);
    }
  }
  __syncthreads();
  if (threadIdx.x < 4) {
    atomicAdd(&stats[threadIdx.x], acc[threadIdx.x]);
    atomicAdd(&stats[4 + threadIdx.x], acc[4 + threadIdx.x]);
  }
}

// ---------------- pos-enc stage 2: LN(global) -> relu -> @ w2 + b2 ----------------
__global__ void pe_stage2_kernel(const float* __restrict__ xin,
                                 const float* __restrict__ stats,
                                 const float* __restrict__ g, const float* __restrict__ be,
                                 const float* __restrict__ w2, const float* __restrict__ b2,
                                 float* __restrict__ delta, int cin) {
  const int t = blockIdx.x * blockDim.x + threadIdx.x;
  if (t >= Bb * Nn * Kk) return;
  const float invcnt = 1.f / (float)(Bb * Nn * Kk);
  float xn[4] = {0.f, 0.f, 0.f, 0.f};
  for (int j = 0; j < cin; ++j) {
    const float mu = stats[j] * invcnt;
    float var = stats[4 + j] * invcnt - mu * mu;
    var = fmaxf(var, 0.f);
    const float z = (xin[(size_t)t * cin + j] - mu) * rsqrtf(var + EPSf) * g[j] + be[j];
    xn[j] = fmaxf(z, 0.f);
  }
  for (int c = 0; c < Cc; ++c) {
    float v = b2[c];
    for (int j = 0; j < cin; ++j) v += xn[j] * w2[j * Cc + c];
    delta[(size_t)t * Cc + c] = v;
  }
}

// ---------------- scatter events into image (atomic add) ----------------
__global__ void scatter_kernel(const float* __restrict__ xytp,
                               const float* __restrict__ features,
                               float* __restrict__ img) {
  const int t = blockIdx.x * blockDim.x + threadIdx.x;
  if (t >= Bb * Nn) return;
  const int b = t / Nn;
  const f32x4 p = *((const f32x4*)xytp + t);
  int yi = (int)roundf(p[2] * Hh); yi = min(max(yi, 0), Hh - 1);
  int xi = (int)roundf(p[1] * Ww); xi = min(max(xi, 0), Ww - 1);
  float* dst = img + (((size_t)b * Hh + yi) * Ww + xi) * CIN2;
  atomicAdd(&dst[0], p[3]);
  atomicAdd(&dst[1], 1.f - p[3]);
  const float* f = features + (size_t)t * Cc;
  for (int c = 0; c < Cc; ++c) atomicAdd(&dst[2 + c], f[c]);
}

// ---------------- im2col (K padded to 864 with zeros) ----------------
__global__ void im2col_kernel(const float* __restrict__ xytp,
                              const float* __restrict__ img,
                              float* __restrict__ col) {
  const size_t t = (size_t)blockIdx.x * blockDim.x + threadIdx.x;
  if (t >= (size_t)Bb * Nn * KCONVP) return;
  const size_t pt = t / KCONVP;
  const int j = (int)(t % KCONVP);
  float v = 0.f;
  if (j < KCONV) {
    const int b = (int)(pt / Nn);
    const f32x4 p = *((const f32x4*)xytp + pt);
    int yi = (int)roundf(p[2] * Hh); yi = min(max(yi, 0), Hh - 1);
    int xi = (int)roundf(p[1] * Ww); xi = min(max(xi, 0), Ww - 1);
    const int fy = j / (FSs * CIN2);
    const int r  = j % (FSs * CIN2);
    const int fx = r / CIN2;
    const int ci = r % CIN2;
    const int y = yi + fy - 2, x = xi + fx - 2;
    if (y >= 0 && y < Hh && x >= 0 && x < Ww)
      v = img[(((size_t)b * Hh + y) * Ww + x) * CIN2 + ci];
  }
  col[t] = v;
}

// ---------------- max-pool psi/alpha over pair_idx neighborhoods ----------------
__global__ void pool_kernel(const float* __restrict__ t_g, const int* __restrict__ pair_idx,
                            float* __restrict__ psi_p, float* __restrict__ alpha_p) {
  const int t = blockIdx.x * blockDim.x + threadIdx.x;
  if (t >= Bb * Nss * Cc) return;
  const int c = t % Cc;
  const int s = (t / Cc) % Nss;
  const int b = t / (Nss * Cc);
  float m1 = -1e30f, m2 = -1e30f;
  for (int k = 0; k < Kk; ++k) {
    const int m = pair_idx[((size_t)b * Nss + s) * Kk + k];
    const float* row = t_g + ((size_t)b * Nn + m) * (3 * Cc);
    m1 = fmaxf(m1, row[Cc + c]);
    m2 = fmaxf(m2, row[2 * Cc + c]);
  }
  psi_p[t] = m1;
  alpha_p[t] = m2;
}

// ---------------- vector attention: one wave32 per point, lane = channel ----------------
__global__ void vec_attn_kernel(const float* __restrict__ varphi,
                                const float* __restrict__ psi, int psiN, int psiStride, int psiOff,
                                const float* __restrict__ alpha, int alStride, int alOff,
                                const int* __restrict__ idx,
                                const float* __restrict__ delta,
                                const float* __restrict__ lg, const float* __restrict__ lb,
                                float* __restrict__ out, int outOff) {
  const int wave = threadIdx.x >> 5;
  const int lane = threadIdx.x & 31;
  const int pt = blockIdx.x * (blockDim.x >> 5) + wave;
  if (pt >= Bb * Nn) return;
  const int b = pt / Nn;
  const float gc = lg[lane], bc = lb[lane];
  const float vq = varphi[(size_t)pt * (3 * Cc) + lane];
  float sc[Kk], av[Kk];
  #pragma unroll
  for (int k = 0; k < Kk; ++k) {
    const int m = idx[(size_t)pt * Kk + k];
    const float pv = psi[((size_t)b * psiN + m) * psiStride + psiOff + lane];
    const float al = alpha[((size_t)b * psiN + m) * alStride + alOff + lane];
    const float dl = delta[((size_t)pt * Kk + k) * Cc + lane];
    const float v = vq - pv + dl;
    const float mean = wave_sum(v) * (1.f / 32.f);
    const float d = v - mean;
    const float var = wave_sum(d * d) * (1.f / 32.f);
    const float zn = d * rsqrtf(var + EPSf) * gc + bc;
    sc[k] = zn * INV_SCALE;
    av[k] = al + dl;
  }
  float mx = sc[0];
  #pragma unroll
  for (int k = 1; k < Kk; ++k) mx = fmaxf(mx, sc[k]);
  float denom = 0.f, acc = 0.f;
  #pragma unroll
  for (int k = 0; k < Kk; ++k) {
    const float e = __expf(sc[k] - mx);
    denom += e;
    acc += e * av[k];
  }
  out[(size_t)pt * (3 * Cc) + outOff + lane] = acc / denom;
}

__global__ void gelu_kernel(float* __restrict__ x, size_t n) {
  const size_t i = (size_t)blockIdx.x * blockDim.x + threadIdx.x;
  if (i < n) {
    const float v = x[i];
    x[i] = 0.5f * v * (1.f + erff(v * 0.70710678118654752f));
  }
}

// =====================================================================
extern "C" void kernel_launch(void* const* d_in, const int* in_sizes, int n_in,
                              void* d_out, int out_size, void* d_ws, size_t ws_size,
                              hipStream_t stream) {
  const float* xytp     = (const float*)d_in[0];
  const float* features = (const float*)d_in[1];
  const float* plw1 = (const float*)d_in[2];  const float* plb1 = (const float*)d_in[3];
  const float* plg  = (const float*)d_in[4];  const float* plbe = (const float*)d_in[5];
  const float* plw2 = (const float*)d_in[6];  const float* plb2 = (const float*)d_in[7];
  const float* lt_w = (const float*)d_in[8];  const float* lt_b = (const float*)d_in[9];
  const float* lnl_g = (const float*)d_in[10]; const float* lnl_b = (const float*)d_in[11];
  const float* psw1 = (const float*)d_in[12]; const float* psb1 = (const float*)d_in[13];
  const float* psg  = (const float*)d_in[14]; const float* psbe = (const float*)d_in[15];
  const float* psw2 = (const float*)d_in[16]; const float* psb2 = (const float*)d_in[17];
  const float* conv_w = (const float*)d_in[18]; const float* conv_b = (const float*)d_in[19];
  const float* lns_g = (const float*)d_in[20]; const float* lns_b = (const float*)d_in[21];
  const float* pgw1 = (const float*)d_in[22]; const float* pgb1 = (const float*)d_in[23];
  const float* pgg  = (const float*)d_in[24]; const float* pgbe = (const float*)d_in[25];
  const float* pgw2 = (const float*)d_in[26]; const float* pgb2 = (const float*)d_in[27];
  const float* gt_w = (const float*)d_in[28]; const float* gt_b = (const float*)d_in[29];
  const float* lng_g = (const float*)d_in[30]; const float* lng_b = (const float*)d_in[31];
  const float* pw1 = (const float*)d_in[32]; const float* pb1 = (const float*)d_in[33];
  const float* pw2 = (const float*)d_in[34]; const float* pb2 = (const float*)d_in[35];

  const int BN  = Bb * Nn;        // 16384
  const int BNK = BN * Kk;        // 262144

  char* ws = (char*)d_ws;
  size_t off = 0;
  auto alloc = [&](size_t bytes) -> void* {
    void* p = ws + off;
    off = (off + bytes + 255) & ~(size_t)255;
    return p;
  };

  int*   idx_l    = (int*)  alloc((size_t)BNK * 4);
  int*   idx_sc   = (int*)  alloc((size_t)BNK * 4);
  int*   inv_pair = (int*)  alloc((size_t)BNK * 4);
  int*   pair_idx = (int*)  alloc((size_t)Bb * Nss * Kk * 4);
  int*   sidx     = (int*)  alloc((size_t)Bb * Nss * 4);
  float* mind     = (float*)alloc((size_t)BN * 4);
  float* s_xytp   = (float*)alloc((size_t)Bb * Nss * 4 * 4);
  float* stats    = (float*)alloc(8 * 4);
  float* pe_tmp   = (float*)alloc((size_t)BNK * 4 * 4);
  float* delta_l  = (float*)alloc((size_t)BNK * Cc * 4);
  float* delta_sc = (float*)alloc((size_t)BNK * Cc * 4);
  float* delta_g  = (float*)alloc((size_t)BNK * Cc * 4);
  float* t_l      = (float*)alloc((size_t)BN * 3 * Cc * 4);
  float* t_g      = (float*)alloc((size_t)BN * 3 * Cc * 4);
  float* st       = (float*)alloc((size_t)BN * 3 * Cc * 4);
  float* img      = (float*)alloc((size_t)Bb * Hh * Ww * CIN2 * 4);
  float* col      = (float*)alloc((size_t)BN * KCONVP * 4);
  float* psi_p    = (float*)alloc((size_t)Bb * Nss * Cc * 4);
  float* alpha_p  = (float*)alloc((size_t)Bb * Nss * Cc * 4);
  float* attn_cat = (float*)alloc((size_t)BN * 3 * Cc * 4);
  float* h1       = (float*)alloc((size_t)BN * Cc * 4);
  // pre-swizzled f16 weights (B-fragment layout)
  _Float16* wz_lt   = (_Float16*)alloc((size_t)1  * 6 * 32 * 16 * 2);
  _Float16* wz_gt   = (_Float16*)alloc((size_t)1  * 6 * 32 * 16 * 2);
  _Float16* wz_conv = (_Float16*)alloc((size_t)27 * 6 * 32 * 16 * 2);
  _Float16* wz_pw1  = (_Float16*)alloc((size_t)3  * 2 * 32 * 16 * 2);
  _Float16* wz_pw2  = (_Float16*)alloc((size_t)1  * 2 * 32 * 16 * 2);
  float* out_f = (float*)d_out;

  auto blocks = [](size_t n, int bs) { return (unsigned)((n + bs - 1) / bs); };
  auto gemm_grid = [](int M) { return (unsigned)((M / 16 + 3) / 4); };  // 4 waves/block

  // ---- weight swizzles ----
  wswz_kernel<<<blocks(1 * 6 * 32, 256), 256, 0, stream>>>(lt_w, wz_lt, Cc, 3 * Cc);
  wswz_kernel<<<blocks(1 * 6 * 32, 256), 256, 0, stream>>>(gt_w, wz_gt, Cc, 3 * Cc);
  wswz_kernel<<<blocks(27 * 6 * 32, 256), 256, 0, stream>>>(conv_w, wz_conv, KCONV, 3 * Cc);
  wswz_kernel<<<blocks(3 * 2 * 32, 256), 256, 0, stream>>>(pw1, wz_pw1, 3 * Cc, Cc);
  wswz_kernel<<<blocks(1 * 2 * 32, 256), 256, 0, stream>>>(pw2, wz_pw2, Cc, Cc);

  // ---- local branch ----
  knn_kernel<<<BN / 256, 256, 0, stream>>>(xytp, Nn, xytp, Nn, idx_l);
  gemm_wmma_kernel<6><<<gemm_grid(BN), 128, 0, stream>>>(features, wz_lt, lt_b, t_l, BN, Cc);
  fill_kernel<<<1, 8, 0, stream>>>(stats, 0.f, 8);
  pe_stage1_kernel<<<blocks(BNK, 256), 256, 0, stream>>>(xytp, nullptr, Nn, idx_l, plw1, plb1, pe_tmp, stats, 0, 4);
  pe_stage2_kernel<<<blocks(BNK, 256), 256, 0, stream>>>(pe_tmp, stats, plg, plbe, plw2, plb2, delta_l, 4);
  vec_attn_kernel<<<blocks(BN * 32, 128), 128, 0, stream>>>(
      t_l, t_l, Nn, 3 * Cc, Cc, t_l, 3 * Cc, 2 * Cc, idx_l, delta_l, lnl_g, lnl_b, attn_cat, 0);

  // ---- spatial (conv) branch ----
  const float r2 = (5.f / Hh) * (5.f / Hh);
  ball_query_kernel<<<blocks(BN, 256), 256, 0, stream>>>(xytp, r2, idx_sc);
  fill_kernel<<<1, 8, 0, stream>>>(stats, 0.f, 8);
  pe_stage1_kernel<<<blocks(BNK, 256), 256, 0, stream>>>(xytp, nullptr, Nn, idx_sc, psw1, psb1, pe_tmp, stats, 1, 2);
  pe_stage2_kernel<<<blocks(BNK, 256), 256, 0, stream>>>(pe_tmp, stats, psg, psbe, psw2, psb2, delta_sc, 2);
  fill_kernel<<<blocks((size_t)Bb * Hh * Ww * CIN2, 256), 256, 0, stream>>>(img, 0.f, (size_t)Bb * Hh * Ww * CIN2);
  scatter_kernel<<<blocks(BN, 256), 256, 0, stream>>>(xytp, features, img);
  im2col_kernel<<<blocks((size_t)BN * KCONVP, 256), 256, 0, stream>>>(xytp, img, col);
  gemm_wmma_kernel<6><<<gemm_grid(BN), 128, 0, stream>>>(col, wz_conv, conv_b, st, BN, KCONVP);
  vec_attn_kernel<<<blocks(BN * 32, 128), 128, 0, stream>>>(
      st, st, Nn, 3 * Cc, Cc, st, 3 * Cc, 2 * Cc, idx_sc, delta_sc, lns_g, lns_b, attn_cat, Cc);

  // ---- global (downsampled) branch ----
  fps_kernel<<<Bb, 256, 0, stream>>>(xytp, mind, sidx);
  gather_sxytp_kernel<<<blocks(Bb * Nss, 256), 256, 0, stream>>>(xytp, sidx, s_xytp);
  knn_kernel<<<(Bb * Nss) / 256, 256, 0, stream>>>(s_xytp, Nss, xytp, Nn, pair_idx);
  knn_kernel<<<BN / 256, 256, 0, stream>>>(xytp, Nn, s_xytp, Nss, inv_pair);
  gemm_wmma_kernel<6><<<gemm_grid(BN), 128, 0, stream>>>(features, wz_gt, gt_b, t_g, BN, Cc);
  fill_kernel<<<1, 8, 0, stream>>>(stats, 0.f, 8);
  pe_stage1_kernel<<<blocks(BNK, 256), 256, 0, stream>>>(xytp, s_xytp, Nss, inv_pair, pgw1, pgb1, pe_tmp, stats, 2, 4);
  pe_stage2_kernel<<<blocks(BNK, 256), 256, 0, stream>>>(pe_tmp, stats, pgg, pgbe, pgw2, pgb2, delta_g, 4);
  pool_kernel<<<blocks(Bb * Nss * Cc, 256), 256, 0, stream>>>(t_g, pair_idx, psi_p, alpha_p);
  vec_attn_kernel<<<blocks(BN * 32, 128), 128, 0, stream>>>(
      t_g, psi_p, Nss, Cc, 0, alpha_p, Cc, 0, inv_pair, delta_g, lng_g, lng_b, attn_cat, 2 * Cc);

  // ---- output MLP ----
  gemm_wmma_kernel<2><<<gemm_grid(BN), 128, 0, stream>>>(attn_cat, wz_pw1, pb1, h1, BN, 3 * Cc);
  gelu_kernel<<<blocks((size_t)BN * Cc, 256), 256, 0, stream>>>(h1, (size_t)BN * Cc);
  gemm_wmma_kernel<2><<<gemm_grid(BN), 128, 0, stream>>>(h1, wz_pw2, pb2, out_f, BN, Cc);

  (void)in_sizes; (void)n_in; (void)out_size; (void)ws_size;
}
